// ShepardsGatedAttentionBase_3393024164496
// MI455X (gfx1250) — matrix operand order repaired and checked by reading, not compiled, not run on hardware
//
#include <hip/hip_runtime.h>
#include <hip/hip_bf16.h>

// ---------------------------------------------------------------------------
// Shepard's gated attention for MI455X (gfx1250, wave32, WMMA bf16,
// async global->LDS tile staging, double-buffered LDS).
//
// Pipeline:
//   1) cvt:      X -> bf16, W_in^T -> bf16, W_out^T -> bf16
//   2) gemm1:    P_bf = bf16(X @ W_in + b_in)           [4096 x 4096]
//   3) sumsq:    q2[h,b,t], k2[h,b,t] from P_bf
//   4) attn:     per (h,b): S = Q K^T (WMMA), w = (eps+dist)^-2,
//                rowsum via WMMA vs ones-B, A = w @ V (WMMA),
//                H_bf = G * A / (eps + rowsum)          [4096 x 1024]
//   5) gemm2:    out = H_bf @ W_out + b_out (f32)       [4096 x 1024]
// ---------------------------------------------------------------------------

typedef __bf16 v16bf __attribute__((ext_vector_type(16)));
typedef __bf16 bf8   __attribute__((ext_vector_type(8)));
typedef float  v8f   __attribute__((ext_vector_type(8)));
typedef unsigned short u16x8 __attribute__((ext_vector_type(8)));

#define EPS_C 1e-4f

__device__ __forceinline__ unsigned short f2bf(float f) {
    unsigned u = __float_as_uint(f);
    unsigned r = u + 0x7FFFu + ((u >> 16) & 1u);   // round-to-nearest-even
    return (unsigned short)(r >> 16);
}
__device__ __forceinline__ float bf2f(unsigned short h) {
    return __uint_as_float(((unsigned)h) << 16);
}

__device__ __forceinline__ v8f wmma_bf16(v16bf a, v16bf b, v8f c) {
    return __builtin_amdgcn_wmma_f32_16x16x32_bf16(
        /*neg_a=*/false, a, /*neg_b=*/false, b,
        /*c_mod=*/(short)0, c, /*reuse_a=*/false, /*reuse_b=*/false);
}

// Async 16B copy global -> LDS (ASYNCcnt-tracked, no data VGPRs).
// LDS destination = LDS base + lds_off; per-lane addresses (ISA 10.x / 15.18.3).
__device__ __forceinline__ void async_ld_b128(const unsigned short* lds_ptr,
                                              const unsigned short* gptr) {
    unsigned lds_off = (unsigned)(unsigned long long)(uintptr_t)lds_ptr;
    unsigned long long ga = (unsigned long long)(uintptr_t)gptr;
    asm volatile("global_load_async_to_lds_b128 %0, %1, off"
                 :: "v"(lds_off), "v"(ga) : "memory");
}
__device__ __forceinline__ void wait_async0() {
    asm volatile("s_wait_asynccnt 0" ::: "memory");
}
// Compiler-only ordering fence (HW keeps same-wave LDS ops in order).
__device__ __forceinline__ void compiler_fence() {
    asm volatile("" ::: "memory");
}

// A-matrix fragment (16x32 bf16): lane m holds K{0..7,16..23},
// lane m+16 holds K{8..15,24..31}  (ISA 7.12.2).
__device__ __forceinline__ v16bf loadA_frag(const unsigned short* base, int stride) {
    int lane = threadIdx.x & 31;
    int row  = lane & 15;
    int coff = (lane < 16) ? 0 : 8;
    const unsigned short* p = base + row * stride + coff;
    bf8 lo = *(const bf8*)(p);
    bf8 hi = *(const bf8*)(p + 16);
    return __builtin_shufflevector(lo, hi, 0,1,2,3,4,5,6,7,8,9,10,11,12,13,14,15);
}

// B-matrix fragment (32x16 bf16) from an N-major (transposed) tile:
// lane n holds K{0..7,16..23}, lane n+16 holds K{8..15,24..31}.
__device__ __forceinline__ v16bf loadB_frag(const unsigned short* base, int stride) {
    int lane = threadIdx.x & 31;
    int row  = lane & 15;
    int coff = (lane < 16) ? 0 : 8;
    const unsigned short* p = base + row * stride + coff;
    bf8 lo = *(const bf8*)(p);
    bf8 hi = *(const bf8*)(p + 16);
    return __builtin_shufflevector(lo, hi, 0,1,2,3,4,5,6,7,8,9,10,11,12,13,14,15);
}

// ---------------------------------------------------------------------------
__global__ void cvt_copy_kernel(const float* __restrict__ in,
                                unsigned short* __restrict__ out, int n) {
    int i = blockIdx.x * blockDim.x + threadIdx.x;
    if (i < n) out[i] = f2bf(in[i]);
}

// in [K][N] row-major -> out [N][K] row-major (bf16)
__global__ void cvt_T_kernel(const float* __restrict__ in,
                             unsigned short* __restrict__ out, int K, int N) {
    long i = (long)blockIdx.x * blockDim.x + threadIdx.x;
    long total = (long)K * N;
    if (i < total) {
        int n = (int)(i / K), k = (int)(i % K);
        out[i] = f2bf(in[(long)k * N + n]);
    }
}

// ---------------------------------------------------------------------------
// C[M,N] = A[M,K] @ Bt[N,K]^T + bias ; A/Bt bf16, accumulate f32.
// Block: 256 threads = 8 waves; tile 128(M) x 64(N), BK = 32.
// Double-buffered LDS fed by async global->LDS copies.
template <bool OUT_F32>
__global__ void __launch_bounds__(256)
gemm_bf16_kernel(const unsigned short* __restrict__ A,
                 const unsigned short* __restrict__ Bt,
                 const float* __restrict__ bias,
                 unsigned short* __restrict__ out_bf,
                 float* __restrict__ out_f,
                 int M, int N, int K) {
    __shared__ __align__(16) unsigned short At[2][128][32];
    __shared__ __align__(16) unsigned short Btt[2][64][32];
    const int bm = blockIdx.x * 128;
    const int bn = blockIdx.y * 64;
    const int tid = threadIdx.x;
    const int wave = tid >> 5, lane = tid & 31;

    const int arow = tid >> 1, ahalf = (tid & 1) * 16;   // A: 16 elems/thread
    const int brow = tid >> 2, bseg = (tid & 3) * 8;     // B: 8 elems/thread

    const v8f vz = {0.f,0.f,0.f,0.f,0.f,0.f,0.f,0.f};
    v8f acc[4] = {vz, vz, vz, vz};

    // Prologue: stage k0 = 0 into buffer 0.
    {
        const unsigned short* asrc = A + (long)(bm + arow) * K + ahalf;
        async_ld_b128(&At[0][arow][ahalf],     asrc);
        async_ld_b128(&At[0][arow][ahalf + 8], asrc + 8);
        const unsigned short* bsrc = Bt + (long)(bn + brow) * K + bseg;
        async_ld_b128(&Btt[0][brow][bseg], bsrc);
    }

    int ping = 0;
    for (int k0 = 0; k0 < K; k0 += 32) {
        wait_async0();        // this wave's copies into buf[ping] done
        __syncthreads();      // all waves done: buf[ping] ready, buf[1-ping] free
        if (k0 + 32 < K) {
            const unsigned short* asrc = A + (long)(bm + arow) * K + (k0 + 32) + ahalf;
            async_ld_b128(&At[ping ^ 1][arow][ahalf],     asrc);
            async_ld_b128(&At[ping ^ 1][arow][ahalf + 8], asrc + 8);
            const unsigned short* bsrc = Bt + (long)(bn + brow) * K + (k0 + 32) + bseg;
            async_ld_b128(&Btt[ping ^ 1][brow][bseg], bsrc);
        }
        v16bf a = loadA_frag(&At[ping][wave * 16][0], 32);
#pragma unroll
        for (int nt = 0; nt < 4; ++nt) {
            v16bf b = loadB_frag(&Btt[ping][nt * 16][0], 32);
            acc[nt] = wmma_bf16(a, b, acc[nt]);
        }
        ping ^= 1;
    }

    // Epilogue. C layout: VGPR r -> M = r (+8 for lanes 16..31), N = lane&15.
    const int rbase = (lane < 16) ? 0 : 8;
    const int mrow  = bm + wave * 16 + rbase;
#pragma unroll
    for (int nt = 0; nt < 4; ++nt) {
        int n = bn + nt * 16 + (lane & 15);
        float bv = bias[n];
#pragma unroll
        for (int r = 0; r < 8; ++r) {
            float v = acc[nt][r] + bv;
            long idx = (long)(mrow + r) * N + n;
            if (OUT_F32) out_f[idx] = v;
            else         out_bf[idx] = f2bf(v);
        }
    }
}

// ---------------------------------------------------------------------------
// q2/k2: sum of squares of Q / K rows from P_bf.  One wave per (h,b,t) row.
__global__ void __launch_bounds__(256)
sumsq_kernel(const unsigned short* __restrict__ Pbf,
             float* __restrict__ q2, float* __restrict__ k2) {
    int tid = threadIdx.x;
    int wave = tid >> 5, lane = tid & 31;
    int row = blockIdx.x * 8 + wave;          // [0, 16*2*2048)
    int t  = row & 2047;
    int hb = row >> 11;                       // h*2 + b
    int b = hb & 1, h = hb >> 1;
    const unsigned short* base = Pbf + ((long)(b * 2048 + t) * 4096) + h * 256;
    float x0 = bf2f(base[lane]),      x1 = bf2f(base[lane + 32]);
    float y0 = bf2f(base[64 + lane]), y1 = bf2f(base[64 + lane + 32]);
    float q = x0 * x0 + x1 * x1;
    float k = y0 * y0 + y1 * y1;
#pragma unroll
    for (int off = 16; off > 0; off >>= 1) {
        q += __shfl_xor(q, off, 32);
        k += __shfl_xor(k, off, 32);
    }
    if (lane == 0) { q2[row] = q; k2[row] = k; }
}

// ---------------------------------------------------------------------------
// Attention: block = 8 waves, each wave owns 16 query rows of one (h,b).
// grid.x = 32 (h,b) * 16 qblocks = 512.  Double-buffered K (async) and
// V^T (software-pipelined vector load + LDS scatter); 1 barrier / chunk.
__global__ void __launch_bounds__(256)
attn_kernel(const unsigned short* __restrict__ Pbf,
            const float* __restrict__ q2g,
            const float* __restrict__ k2g,
            unsigned short* __restrict__ Hbf) {
    __shared__ __align__(16) unsigned short Ktile[2][32][64];   // [s][d]
    __shared__ __align__(16) unsigned short VtileT[2][64][32];  // [e][s]
    __shared__ __align__(16) unsigned short Wst[8][16][32];     // per-wave [m][s]

    const int tid = threadIdx.x, wave = tid >> 5, lane = tid & 31;
    const int qblk = blockIdx.x & 15;
    const int hb   = blockIdx.x >> 4;          // h*2 + b
    const int b = hb & 1, h = hb >> 1;
    const int qtile = qblk * 128 + wave * 16;

    const int srow = tid >> 3, seg = (tid & 7) * 8;   // staging roles

    // Q fragments (d 0..31 and 32..63) straight from global (A layout).
    v16bf qf0, qf1;
    {
        int row  = lane & 15;
        int coff = (lane < 16) ? 0 : 8;
        const unsigned short* p =
            Pbf + (long)(b * 2048 + qtile + row) * 4096 + h * 256 + coff;
        bf8 lo0 = *(const bf8*)(p);      bf8 hi0 = *(const bf8*)(p + 16);
        bf8 lo1 = *(const bf8*)(p + 32); bf8 hi1 = *(const bf8*)(p + 48);
        qf0 = __builtin_shufflevector(lo0, hi0, 0,1,2,3,4,5,6,7,8,9,10,11,12,13,14,15);
        qf1 = __builtin_shufflevector(lo1, hi1, 0,1,2,3,4,5,6,7,8,9,10,11,12,13,14,15);
    }
    float q2r[8];
    {
        const float* p = q2g + (long)hb * 2048 + qtile + ((lane < 16) ? 0 : 8);
#pragma unroll
        for (int r = 0; r < 8; ++r) q2r[r] = p[r];
    }
    const __bf16 one = __builtin_bit_cast(__bf16, (unsigned short)0x3F80);
    const v16bf ones = {one,one,one,one,one,one,one,one,
                        one,one,one,one,one,one,one,one};

    const v8f vz = {0.f,0.f,0.f,0.f,0.f,0.f,0.f,0.f};
    v8f accA[4] = {vz, vz, vz, vz};
    v8f accS = vz;

    const unsigned short* kvbase = Pbf + (long)(b * 2048) * 4096 + h * 256;
    const float* k2base = k2g + (long)hb * 2048;

    // Prologue: async K chunk 0 into buf0; V chunk 0 into registers.
    async_ld_b128(&Ktile[0][srow][seg], kvbase + (long)srow * 4096 + 64 + seg);
    u16x8 vv = *(const u16x8*)(kvbase + (long)srow * 4096 + 128 + seg);

    int ping = 0;
    for (int sb = 0; sb < 2048; sb += 32) {
        wait_async0();                 // K copy into buf[ping] complete (this wave)
        {   // scatter this chunk's V rows into transposed tile
#pragma unroll
            for (int j = 0; j < 8; ++j) VtileT[ping][seg + j][srow] = vv[j];
        }
        __syncthreads();               // buf[ping] (K async + V scatter) visible

        if (sb + 32 < 2048) {          // stage next chunk (overlaps the WMMAs)
            const long roff = (long)(sb + 32 + srow) * 4096;
            async_ld_b128(&Ktile[ping ^ 1][srow][seg], kvbase + roff + 64 + seg);
            vv = *(const u16x8*)(kvbase + roff + 128 + seg);
        }

        // S = Q K^T for 16q x 32s (two 16x16 C tiles, K = 64 in two steps)
        v8f s0 = vz, s1 = vz;
        {
            v16bf b0 = loadB_frag(&Ktile[ping][0][0], 64);    // s 0..15,  d 0..31
            v16bf b1 = loadB_frag(&Ktile[ping][16][0], 64);   // s 16..31, d 0..31
            s0 = wmma_bf16(qf0, b0, s0);
            s1 = wmma_bf16(qf0, b1, s1);
            v16bf b2 = loadB_frag(&Ktile[ping][0][32], 64);   // s 0..15,  d 32..63
            v16bf b3 = loadB_frag(&Ktile[ping][16][32], 64);  // s 16..31, d 32..63
            s0 = wmma_bf16(qf1, b2, s0);
            s1 = wmma_bf16(qf1, b3, s1);
        }
        {   // Shepard weights, staged into A layout (per-wave region)
            float k2a = k2base[sb + (lane & 15)];
            float k2b = k2base[sb + 16 + (lane & 15)];
            int rbase = (lane < 16) ? 0 : 8;
            int col = lane & 15;
#pragma unroll
            for (int r = 0; r < 8; ++r) {
                float d2a = q2r[r] + k2a - 2.0f * s0[r];
                float ta  = EPS_C + sqrtf(fmaxf(d2a, 0.0f));
                float wa  = 1.0f / (ta * ta);
                float d2b = q2r[r] + k2b - 2.0f * s1[r];
                float tb  = EPS_C + sqrtf(fmaxf(d2b, 0.0f));
                float wb  = 1.0f / (tb * tb);
                Wst[wave][rbase + r][col]      = f2bf(wa);
                Wst[wave][rbase + r][16 + col] = f2bf(wb);
            }
        }
        // Same-wave LDS RAW: HW keeps LDS ops in order; stop compiler reordering.
        compiler_fence();

        v16bf wf = loadA_frag(&Wst[wave][0][0], 32);
        accS = wmma_bf16(wf, ones, accS);   // rowsum (every column == sum_s w)
#pragma unroll
        for (int nt = 0; nt < 4; ++nt) {
            v16bf vf = loadB_frag(&VtileT[ping][nt * 16][0], 32);
            accA[nt] = wmma_bf16(wf, vf, accA[nt]);
        }
        ping ^= 1;
    }

    // Normalize, gate, store H (bf16).
    {
        int rbase = (lane < 16) ? 0 : 8;
        int col = lane & 15;
#pragma unroll
        for (int r = 0; r < 8; ++r) {
            float denom = 1.0f / (EPS_C + accS[r]);
            int row = qtile + rbase + r;
            const unsigned short* gsrc =
                Pbf + (long)(b * 2048 + row) * 4096 + h * 256 + 192;
            unsigned short* hdst =
                Hbf + (long)(b * 2048 + row) * 1024 + h * 64;
#pragma unroll
            for (int nt = 0; nt < 4; ++nt) {
                int e = nt * 16 + col;
                float a = accA[nt][r] * denom;
                float g = bf2f(gsrc[e]);
                hdst[e] = f2bf(g * a);
            }
        }
    }
}

// ---------------------------------------------------------------------------
extern "C" void kernel_launch(void* const* d_in, const int* in_sizes, int n_in,
                              void* d_out, int out_size, void* d_ws, size_t ws_size,
                              hipStream_t stream) {
    const float* X     = (const float*)d_in[0];   // [2,2048,1024]
    const float* W_in  = (const float*)d_in[1];   // [1024,4096]
    const float* b_in  = (const float*)d_in[2];   // [4096]
    const float* W_out = (const float*)d_in[3];   // [1024,1024]
    const float* b_out = (const float*)d_in[4];   // [1024]
    float* out = (float*)d_out;                   // [2,2048,1024] f32

    char* ws = (char*)d_ws;
    unsigned short* Xbf   = (unsigned short*)(ws);                        // 8 MB
    unsigned short* WinT  = (unsigned short*)(ws + (size_t)( 8u << 20)); // 8 MB
    unsigned short* WoutT = (unsigned short*)(ws + (size_t)(16u << 20)); // 2 MB
    unsigned short* Pbf   = (unsigned short*)(ws + (size_t)(18u << 20)); // 32 MB
    float*          q2    = (float*)        (ws + (size_t)(50u << 20));  // 256 KB
    float*          k2    = (float*)        (ws + (size_t)(51u << 20));  // 256 KB
    unsigned short* Hbf   = (unsigned short*)(ws + (size_t)(52u << 20)); // 8 MB

    cvt_copy_kernel<<<16384, 256, 0, stream>>>(X, Xbf, 4096 * 1024);
    cvt_T_kernel<<<16384, 256, 0, stream>>>(W_in, WinT, 1024, 4096);
    cvt_T_kernel<<<4096, 256, 0, stream>>>(W_out, WoutT, 1024, 1024);

    gemm_bf16_kernel<false><<<dim3(32, 64), 256, 0, stream>>>(
        Xbf, WinT, b_in, Pbf, nullptr, 4096, 4096, 1024);

    sumsq_kernel<<<8192, 256, 0, stream>>>(Pbf, q2, k2);
    attn_kernel<<<512, 256, 0, stream>>>(Pbf, q2, k2, Hbf);

    gemm_bf16_kernel<true><<<dim3(32, 16), 256, 0, stream>>>(
        Hbf, WoutT, b_out, nullptr, out, 4096, 1024, 1024);
}
